// AttentionHead_Latex_withMask_36429912605013
// MI455X (gfx1250) — compile-verified, matrix operand order checked by reading, not run on hardware
//
#include <hip/hip_runtime.h>

#define BB 8
#define TT 4096
#define CDIM 128
#define HSD 64
#define KPAD 72   // 64 + 8 halfwords: 144B rows -> conflict-free 16-lane bursts

// Async global->LDS staging path (gfx1250): guarded so the kernel compiles on
// toolchains without the builtins (falls back to a register bounce).
#if defined(__has_builtin)
#  if __has_builtin(__builtin_amdgcn_global_load_async_to_lds_b128) && \
      __has_builtin(__builtin_amdgcn_s_wait_asynccnt)
#    define HAVE_ASYNC_LDS 1
#  endif
#endif
#ifndef HAVE_ASYNC_LDS
#  define HAVE_ASYNC_LDS 0
#endif

typedef __attribute__((ext_vector_type(16))) _Float16 v16h;
typedef __attribute__((ext_vector_type(8)))  _Float16 v8h;
typedef __attribute__((ext_vector_type(8)))  float    v8f;
typedef __attribute__((ext_vector_type(2)))  float    v2f;
typedef __attribute__((ext_vector_type(4)))  int      v4i;

// Build a 16-bit A/B WMMA fragment from a row pointer.
// Per-lane layout for 16-bit 16x32 operands (CDNA5 ISA 7.12.2):
//   lanes 0-15 : K in {0..7, 16..23}, lanes 16-31 : K in {8..15, 24..31},
// so with p already offset by (half*8), elements are p[0..7] and p[16..23].
__device__ __forceinline__ v16h load_frag(const _Float16* p) {
    v8h lo = *(const v8h*)p;
    v8h hi = *(const v8h*)(p + 16);
    v16h r;
#pragma unroll
    for (int i = 0; i < 8; ++i) { r[i] = lo[i]; r[i + 8] = hi[i]; }
    return r;
}

__device__ __forceinline__ v8f wmma_f16(v16h a, v16h b, v8f c) {
    return __builtin_amdgcn_wmma_f32_16x16x32_f16(
        /*neg_a=*/false, a, /*neg_b=*/false, b,
        /*c_mod=*/(short)0, c, /*reuse_a=*/false, /*reuse_b=*/false);
}

// Copy 16 bytes global->LDS, async DMA (ASYNCcnt-tracked) when available.
__device__ __forceinline__ void stage16(const _Float16* g, _Float16* l) {
#if HAVE_ASYNC_LDS
    typedef __attribute__((address_space(1))) v4i* gv4i_t;   // global src
    typedef __attribute__((address_space(3))) v4i* lv4i_t;   // LDS dst
    __builtin_amdgcn_global_load_async_to_lds_b128(
        (gv4i_t)g, (lv4i_t)l, /*offset=*/0, /*cpol=*/0);
#else
    *(v8h*)l = *(const v8h*)g;
#endif
}

// Wait until at most N async transfers remain in flight (in-order completion:
// ASYNCcnt<=8 with 16 outstanding proves the older buffer's 8 are done).
template <int N>
__device__ __forceinline__ void stage_wait() {
#if HAVE_ASYNC_LDS
    __builtin_amdgcn_s_wait_asynccnt(N);
#endif
}

// ---------------------------------------------------------------------------
// Kernel 1: fused QKV projection + RoPE + scale, fp32 -> f16.
// One wave per (b,t); lane i owns head-dim pair (2i, 2i+1).
// Q is pre-scaled by log2(e)/sqrt(C) so softmax runs in the exp2 domain.
// V is stored transposed [B][HS][T] so PV B-fragments are contiguous in K.
// ---------------------------------------------------------------------------
__global__ __launch_bounds__(32)
void proj_rope_kernel(const float* __restrict__ x,
                      const float* __restrict__ Wq,
                      const float* __restrict__ Wk,
                      const float* __restrict__ Wv,
                      const float* __restrict__ fc,
                      const float* __restrict__ fs,
                      _Float16* __restrict__ Qg,
                      _Float16* __restrict__ Kg,
                      _Float16* __restrict__ Vt) {
    const int bt = blockIdx.x;
    const int b  = bt / TT;
    const int t  = bt - b * TT;
    const int i  = threadIdx.x;            // pair index 0..31

    const float* xr = x + (size_t)bt * CDIM;
    float q0 = 0.f, q1 = 0.f, k0 = 0.f, k1 = 0.f, p0 = 0.f, p1 = 0.f;
#pragma unroll 4
    for (int c = 0; c < CDIM; ++c) {
        float xv = xr[c];                                   // wave-uniform
        v2f wq = *(const v2f*)(Wq + (size_t)c * HSD + 2 * i);
        v2f wk = *(const v2f*)(Wk + (size_t)c * HSD + 2 * i);
        v2f wv = *(const v2f*)(Wv + (size_t)c * HSD + 2 * i);
        q0 = fmaf(xv, wq[0], q0); q1 = fmaf(xv, wq[1], q1);
        k0 = fmaf(xv, wk[0], k0); k1 = fmaf(xv, wk[1], k1);
        p0 = fmaf(xv, wv[0], p0); p1 = fmaf(xv, wv[1], p1);
    }
    const float cs = fc[(size_t)t * (HSD / 2) + i];
    const float sn = fs[(size_t)t * (HSD / 2) + i];
    // 1/sqrt(128) * log2(e): scores land directly in the exp2 domain.
    const float qscale = 0.08838834764831845f * 1.4426950408889634f;
    float qa = (q0 * cs - q1 * sn) * qscale;
    float qb = (q0 * sn + q1 * cs) * qscale;
    float ka = k0 * cs - k1 * sn;
    float kb = k0 * sn + k1 * cs;

    const size_t base = (size_t)bt * HSD + 2 * i;
    Qg[base]     = (_Float16)qa;
    Qg[base + 1] = (_Float16)qb;
    Kg[base]     = (_Float16)ka;
    Kg[base + 1] = (_Float16)kb;
    Vt[((size_t)b * HSD + 2 * i)     * TT + t] = (_Float16)p0;
    Vt[((size_t)b * HSD + 2 * i + 1) * TT + t] = (_Float16)p1;
}

// ---------------------------------------------------------------------------
// Issue the 8 per-thread async transfers staging one 64-key K/V block.
// ---------------------------------------------------------------------------
__device__ __forceinline__ void stage_block(
    const _Float16* __restrict__ Kg, const _Float16* __restrict__ Vt,
    int b, int kb, int tid,
    _Float16 (* __restrict__ KsB)[KPAD], _Float16 (* __restrict__ VsB)[KPAD]) {
#pragma unroll
    for (int j = 0; j < 4; ++j) {
        const int c   = tid + 128 * j;     // 0..511: 16B chunk id
        const int row = c >> 3;
        const int col = (c & 7) * 8;
        stage16(Kg + ((size_t)b * TT + kb + row) * HSD + col, &KsB[row][col]);
        stage16(Vt + ((size_t)b * HSD + row) * TT + kb + col, &VsB[row][col]);
    }
}

// ---------------------------------------------------------------------------
// One 64-key block for a 16-row Q tile owned by one wave, fed from the
// block-shared LDS K/V tiles. CAUSAL=false for blocks fully below the
// diagonal (padding mask still applied). 18 WMMA per call.
// ---------------------------------------------------------------------------
template <bool CAUSAL>
__device__ __forceinline__ void process_block(
    int kb, int qbase, int hh, int lr,
    const _Float16 (* __restrict__ Ks)[KPAD],
    const _Float16 (* __restrict__ Vs)[KPAD],
    const int* __restrict__ mrowptr,
    const v16h& aq0, const v16h& aq1, const v16h& ones,
    v8f& o0, v8f& o1, v8f& o2, v8f& o3,
    float* __restrict__ mrow, float* __restrict__ lrow,
    _Float16 (* __restrict__ plds)[64]) {

    // --- S = Q * K^T: 4 key subtiles x 2 k-parts (K tile from LDS)
    v8f st[4];
#pragma unroll
    for (int s4 = 0; s4 < 4; ++s4) {
        const _Float16* kr = &Ks[16 * s4 + lr][hh * 8];
        v16h bk0 = load_frag(kr);
        v16h bk1 = load_frag(kr + 32);
        v8f acc = {};
        acc = wmma_f16(aq0, bk0, acc);
        acc = wmma_f16(aq1, bk1, acc);
        st[s4] = acc;
    }

    int pad[4];
#pragma unroll
    for (int t = 0; t < 4; ++t) pad[t] = mrowptr[kb + 16 * t + lr];

    // --- online softmax (C-layout: VGPR r holds row r+8*hh, column lr)
#pragma unroll
    for (int r = 0; r < 8; ++r) {
        const int qi = qbase + r + hh * 8;
        float xv[4];
#pragma unroll
        for (int t = 0; t < 4; ++t) {
            bool ok = (pad[t] != 0);
            if (CAUSAL) ok = ok && ((kb + 16 * t + lr) <= qi);
            xv[t] = ok ? st[t][r] : -1e30f;
        }
        float cm = fmaxf(fmaxf(xv[0], xv[1]), fmaxf(xv[2], xv[3]));
#pragma unroll
        for (int off = 1; off < 16; off <<= 1)
            cm = fmaxf(cm, __shfl_xor(cm, off, 16));
        const float newm = fmaxf(mrow[r], cm);
        const float corr = exp2f(mrow[r] - newm);
        mrow[r] = newm;
        lrow[r] *= corr;
        o0[r] *= corr; o1[r] *= corr; o2[r] *= corr; o3[r] *= corr;
#pragma unroll
        for (int t = 0; t < 4; ++t) {
            float e = exp2f(xv[t] - newm);
            plds[r + hh * 8][16 * t + lr] = (_Float16)e;
        }
    }

    // Per-wave LDS region; same-wave DS ops are in order, compiler inserts
    // the s_wait_dscnt before the dependent A-fragment reads.
    v16h ap0 = load_frag(&plds[lr][hh * 8]);
    v16h ap1 = load_frag(&plds[lr][32 + hh * 8]);

    // --- rowsum(P) via WMMA against ones(64x16): lands in C-layout == lrow.
    v8f srow = {};
    srow = wmma_f16(ap0, ones, srow);
    srow = wmma_f16(ap1, ones, srow);
#pragma unroll
    for (int r = 0; r < 8; ++r) lrow[r] += srow[r];

    // --- O += P * V: 4 head-dim chunks x 2 k-parts (V tile from LDS)
#pragma unroll
    for (int h = 0; h < 4; ++h) {
        const _Float16* vr = &Vs[h * 16 + lr][hh * 8];
        v16h bv0 = load_frag(vr);
        v16h bv1 = load_frag(vr + 32);
        if (h == 0)      { o0 = wmma_f16(ap0, bv0, o0); o0 = wmma_f16(ap1, bv1, o0); }
        else if (h == 1) { o1 = wmma_f16(ap0, bv0, o1); o1 = wmma_f16(ap1, bv1, o1); }
        else if (h == 2) { o2 = wmma_f16(ap0, bv0, o2); o2 = wmma_f16(ap1, bv1, o2); }
        else             { o3 = wmma_f16(ap0, bv0, o3); o3 = wmma_f16(ap1, bv1, o3); }
    }
}

// ---------------------------------------------------------------------------
// Kernel 2: flash attention, double-buffered async staging.
// Block = 4 waves, each wave owns 16 query rows. All waves share the staged
// 64-key K/V LDS tiles; trip count nkb = blockIdx.x+1 is identical for every
// wave, so barriers are uniform and EXEC stays all-ones at each WMMA.
// ---------------------------------------------------------------------------
__global__ __launch_bounds__(128)
void flash_attn_kernel(const _Float16* __restrict__ Qg,
                       const _Float16* __restrict__ Kg,
                       const _Float16* __restrict__ Vt,
                       const int* __restrict__ mask,
                       float* __restrict__ out) {
    __shared__ _Float16 Ks[2][64][KPAD];   // K blocks: row=key, col=head dim
    __shared__ _Float16 Vs[2][64][KPAD];   // V blocks: row=dim, col=key
    __shared__ _Float16 plds[4][16][64];   // per-wave P tile

    const int tid  = threadIdx.x;
    const int wave = tid >> 5;
    const int lane = tid & 31;
    const int hh   = lane >> 4;            // half-wave
    const int lr   = lane & 15;            // lane within half
    const int b    = blockIdx.y;
    const int qbase = blockIdx.x * 64 + wave * 16;

    // Issue block 0's DMA before anything else so it overlaps the Q loads.
    stage_block(Kg, Vt, b, 0, tid, Ks[0], Vs[0]);

    // Q A-fragments: lane holds row (qbase+lr), k-split by half-wave.
    const _Float16* qrow = Qg + ((size_t)b * TT + qbase + lr) * HSD + hh * 8;
    const v16h aq0 = load_frag(qrow);
    const v16h aq1 = load_frag(qrow + 32);

    v16h ones;
#pragma unroll
    for (int i = 0; i < 16; ++i) ones[i] = (_Float16)1.0f;

    v8f o0 = {}, o1 = {}, o2 = {}, o3 = {};
    float mrow[8], lrow[8];
#pragma unroll
    for (int r = 0; r < 8; ++r) { mrow[r] = -1e30f; lrow[r] = 0.f; }

    const int* mrowptr = mask + (size_t)b * TT;
    // ceil((qbase+16)/64) == blockIdx.x+1 for all four waves.
    const int nkb = blockIdx.x + 1;

    for (int blk = 0; blk < nkb; ++blk) {
        const int kb  = blk * 64;
        const int cur = blk & 1;

        if (blk + 1 < nkb) {
            // Issue next block's DMA, then wait only for the current buffer:
            // async loads complete in order, so ASYNCcnt<=8 (of 16 in flight)
            // proves the older 8 transfers have landed.
            stage_block(Kg, Vt, b, kb + 64, tid, Ks[1 - cur], Vs[1 - cur]);
            stage_wait<8>();
        } else {
            stage_wait<0>();
        }
        __syncthreads();                   // current tiles visible to all waves

        if (kb + 128 < TT) {               // warm L2 ahead of the DMA engine
            __builtin_prefetch(
                (const void*)(Kg + ((size_t)b * TT + kb + 128 + lane) * HSD), 0, 3);
            __builtin_prefetch(
                (const void*)(Vt + ((size_t)b * HSD + lane) * TT + kb + 128), 0, 3);
            __builtin_prefetch(
                (const void*)(Vt + ((size_t)b * HSD + 32 + lane) * TT + kb + 128), 0, 3);
        }

        // Wave-uniform at runtime; readfirstlane makes it an SGPR so the
        // compiler emits a scalar branch instead of exec masking.
        if (__builtin_amdgcn_readfirstlane((int)(kb + 63 <= qbase)))
            process_block<false>(kb, qbase, hh, lr, Ks[cur], Vs[cur], mrowptr,
                                 aq0, aq1, ones, o0, o1, o2, o3,
                                 mrow, lrow, plds[wave]);
        else
            process_block<true>(kb, qbase, hh, lr, Ks[cur], Vs[cur], mrowptr,
                                aq0, aq1, ones, o0, o1, o2, o3,
                                mrow, lrow, plds[wave]);
        __syncthreads();                   // tiles consumed; buffer reusable
    }

    // --- epilogue: normalize and store fp32 output [B][T][HS]
#pragma unroll
    for (int r = 0; r < 8; ++r) {
        const float inv = 1.0f / lrow[r];
        const size_t ro = ((size_t)b * TT + qbase + r + hh * 8) * HSD;
        out[ro + 0  + lr] = o0[r] * inv;
        out[ro + 16 + lr] = o1[r] * inv;
        out[ro + 32 + lr] = o2[r] * inv;
        out[ro + 48 + lr] = o3[r] * inv;
    }
}

// ---------------------------------------------------------------------------
extern "C" void kernel_launch(void* const* d_in, const int* in_sizes, int n_in,
                              void* d_out, int out_size, void* d_ws, size_t ws_size,
                              hipStream_t stream) {
    const float* x    = (const float*)d_in[0];
    const float* Wq   = (const float*)d_in[1];
    const float* Wk   = (const float*)d_in[2];
    const float* Wv   = (const float*)d_in[3];
    const float* fc   = (const float*)d_in[4];
    const float* fs   = (const float*)d_in[5];
    const int*   mask = (const int*)d_in[6];
    float* out = (float*)d_out;

    const size_t qkv = (size_t)BB * TT * HSD;   // 4 MiB f16 each
    _Float16* Qg = (_Float16*)d_ws;
    _Float16* Kg = Qg + qkv;
    _Float16* Vt = Kg + qkv;

    proj_rope_kernel<<<BB * TT, 32, 0, stream>>>(x, Wq, Wk, Wv, fc, fs, Qg, Kg, Vt);

    dim3 grid(TT / 64, BB);
    flash_attn_kernel<<<grid, 128, 0, stream>>>(Qg, Kg, Vt, mask, out);
}
